// RGLRU_27401891349118
// MI455X (gfx1250) — matrix operand order repaired and checked by reading, not compiled, not run on hardware
//
#include <hip/hip_runtime.h>
#include <hip/hip_bf16.h>

typedef __attribute__((ext_vector_type(16))) __bf16 v16bf;
typedef __attribute__((ext_vector_type(8)))  float  v8f;

#define B_  8
#define T_  4096
#define D_  1024
#define H_  8
#define TC  64          // time rows per GEMM workgroup
#define CL  512         // scan chunk length
#define NC  (T_ / CL)   // 8 chunks

// ---------------------------------------------------------------------------
// Kernel 1: transpose + convert weights to bf16, layout [g][h][n][k]
// ---------------------------------------------------------------------------
__global__ __launch_bounds__(256)
void rglru_prep_w(const float* __restrict__ ig_w,
                  const float* __restrict__ ag_w,
                  __bf16* __restrict__ wt) {
  int tid = blockIdx.x * 256 + threadIdx.x;   // 0 .. 262143
  int g  = tid >> 17;
  int r  = tid & 131071;
  int h  = r >> 14;
  int nk = r & 16383;
  int n  = nk >> 7;
  int k  = nk & 127;
  const float* w = (g == 0) ? ig_w : ag_w;
  wt[tid] = (__bf16)w[(h << 14) + (k << 7) + n];
}

// ---------------------------------------------------------------------------
// Kernel 2: fused block-diag GEMMs (bf16 WMMA) + gates + pointwise.
// Both weight matrices resident in LDS; A fragment shared by both gates.
// Writes a -> a_ws, normed_x -> outp (in place, scanned later).
// ---------------------------------------------------------------------------
__global__ __launch_bounds__(256)
void rglru_gates(const float* __restrict__ x,
                 const float* __restrict__ ig_b,
                 const float* __restrict__ ag_b,
                 const float* __restrict__ a_param,
                 const __bf16* __restrict__ wt,
                 float* __restrict__ a_ws,
                 float* __restrict__ outp) {
  __shared__ __bf16 xbf[TC * 128];        // 16 KB: x tile, bf16, [m][k]
  __shared__ __bf16 wlds[2][128 * 128];   // 64 KB: both weight matrices, [n][k]

  const int tid   = threadIdx.x;
  const int wgid  = blockIdx.x;                 // 4096 = B * (T/TC) * H
  const int h     = wgid & 7;
  const int chunk = (wgid >> 3) & 63;
  const int b     = wgid >> 9;
  const int t0    = chunk * TC;
  const int lane  = tid & 31;
  const int wave  = tid >> 5;
  const int l15   = lane & 15;
  const bool hi   = lane >= 16;
  const int n0    = wave * 16;                  // this wave's N-tile

  // ---- stage x tile (f32 -> bf16) into LDS ----
  const size_t xbase = ((size_t)b * T_ + t0) * D_ + h * 128;
  for (int i = tid; i < TC * 128 / 4; i += 256) {
    int e = i * 4;
    int row = e >> 7, col = e & 127;
    float4 v = *(const float4*)(x + xbase + (size_t)row * D_ + col);
    __bf16* o = &xbf[row * 128 + col];
    o[0] = (__bf16)v.x; o[1] = (__bf16)v.y;
    o[2] = (__bf16)v.z; o[3] = (__bf16)v.w;
  }

  // ---- stage both weight matrices (bf16, [n][k]) into LDS ----
  {
    const uint4* w0 = (const uint4*)(wt + ((size_t)h << 14));              // ig
    const uint4* w1 = (const uint4*)(wt + (((size_t)H_ + h) << 14));       // ag
    uint4* d0 = (uint4*)&wlds[0][0];
    uint4* d1 = (uint4*)&wlds[1][0];
    for (int i = tid; i < 2048; i += 256) {
      d0[i] = w0[i];
      d1[i] = w1[i];
    }
  }
  __syncthreads();

  v8f acc[2][4];
#pragma unroll
  for (int g = 0; g < 2; ++g)
#pragma unroll
    for (int mt = 0; mt < 4; ++mt)
#pragma unroll
      for (int e = 0; e < 8; ++e) acc[g][mt][e] = 0.0f;

  const int khalf = hi ? 16 : 0;   // A: high half-wave holds K+16 block
  const int koffB = hi ? 16 : 0;   // B: high half-wave holds K+16 block

#pragma unroll
  for (int ks = 0; ks < 4; ++ks) {
    // B fragments for both gates: 32x16, lane = column n0+l15, 16 consecutive K
    union { uint4 u[2]; v16bf v; } Bf0, Bf1;
    const int boff = (n0 + l15) * 128 + ks * 32 + koffB;
    Bf0.u[0] = *(const uint4*)&wlds[0][boff];
    Bf0.u[1] = *(const uint4*)&wlds[0][boff + 8];
    Bf1.u[0] = *(const uint4*)&wlds[1][boff];
    Bf1.u[1] = *(const uint4*)&wlds[1][boff + 8];
#pragma unroll
    for (int mt = 0; mt < 4; ++mt) {
      // A fragment: 16x32, lane row = l15, 16 consecutive K [khalf..+15]
      union { uint4 u[2]; v16bf v; } Af;
      const __bf16* ap = &xbf[(mt * 16 + l15) * 128 + ks * 32 + khalf];
      Af.u[0] = *(const uint4*)ap;
      Af.u[1] = *(const uint4*)(ap + 8);
      acc[0][mt] = __builtin_amdgcn_wmma_f32_16x16x32_bf16(
          false, Af.v, false, Bf0.v, (short)0, acc[0][mt], false, false);
      acc[1][mt] = __builtin_amdgcn_wmma_f32_16x16x32_bf16(
          false, Af.v, false, Bf1.v, (short)0, acc[1][mt], false, false);
    }
  }

  // ---- pointwise epilogue ----
  const int nloc = n0 + l15;
  const int d    = h * 128 + nloc;
  const float bi  = ig_b[h * 128 + nloc];
  const float ba  = ag_b[h * 128 + nloc];
  const float apv = a_param[d];
  const float sp  = __logf(1.0f + __expf(apv));     // softplus(a_param)
  const int mbase = hi ? 8 : 0;

#pragma unroll
  for (int mt = 0; mt < 4; ++mt) {
#pragma unroll
    for (int r = 0; r < 8; ++r) {
      int m = mt * 16 + mbase + r;                  // C layout: M=r / M=8+r
      int t = t0 + m;
      float gx = 1.0f / (1.0f + __expf(-(acc[0][mt][r] + bi)));
      float ga = 1.0f / (1.0f + __expf(-(acc[1][mt][r] + ba)));
      float la = -8.0f * ga * sp;
      float av = __expf(la);
      float mult = __builtin_sqrtf(fmaxf(1.0f - av * av, 1e-7f));
      size_t oidx = ((size_t)b * T_ + t) * D_ + d;
      float xv = x[oidx];
      a_ws[oidx] = av;
      outp[oidx] = xv * gx * mult;                  // normed_x
    }
  }
}

// ---------------------------------------------------------------------------
// Kernel 3: per-chunk reduce: P = prod a, S = local scan final (h0 = 0)
// ---------------------------------------------------------------------------
__global__ __launch_bounds__(256)
void rglru_scan1(const float* __restrict__ a_ws, const float* __restrict__ nx,
                 float* __restrict__ P, float* __restrict__ S) {
  int tid = blockIdx.x * 256 + threadIdx.x;   // 65536 = NC*B*D
  int d = tid & (D_ - 1);
  int b = (tid >> 10) & 7;
  int c = tid >> 13;
  size_t idx = ((size_t)b * T_ + c * CL) * D_ + d;
  float p = 1.0f, s = 0.0f;
  for (int t = 0; t < CL; ++t) {
    if ((t & 15) == 0) {
      __builtin_prefetch(a_ws + idx + (size_t)16 * D_, 0, 0);
      __builtin_prefetch(nx   + idx + (size_t)16 * D_, 0, 0);
    }
    float av = a_ws[idx];
    float w  = nx[idx];
    s = fmaf(av, s, w);
    p *= av;
    idx += D_;
  }
  P[tid] = p;
  S[tid] = s;
}

// ---------------------------------------------------------------------------
// Kernel 4: tiny carry scan over chunks: C[c] = h at start of chunk c
// ---------------------------------------------------------------------------
__global__ __launch_bounds__(256)
void rglru_scan2(const float* __restrict__ P, const float* __restrict__ S,
                 float* __restrict__ C) {
  int tid = blockIdx.x * 256 + threadIdx.x;   // 8192 = B*D
  int d = tid & (D_ - 1);
  int b = tid >> 10;
  float carry = 0.0f;
#pragma unroll
  for (int c = 0; c < NC; ++c) {
    int i = (c << 13) + (b << 10) + d;
    C[i] = carry;
    carry = fmaf(P[i], carry, S[i]);
  }
}

// ---------------------------------------------------------------------------
// Kernel 5: apply carries, in-place scan on outp (normed_x -> h)
// ---------------------------------------------------------------------------
__global__ __launch_bounds__(256)
void rglru_scan3(const float* __restrict__ a_ws, const float* __restrict__ C,
                 float* __restrict__ outp) {
  int tid = blockIdx.x * 256 + threadIdx.x;   // 65536
  int d = tid & (D_ - 1);
  int b = (tid >> 10) & 7;
  int c = tid >> 13;
  size_t idx = ((size_t)b * T_ + c * CL) * D_ + d;
  float hh = C[tid];
  for (int t = 0; t < CL; ++t) {
    if ((t & 15) == 0) {
      __builtin_prefetch(a_ws + idx + (size_t)16 * D_, 0, 0);
      __builtin_prefetch(outp + idx + (size_t)16 * D_, 0, 0);
    }
    float av = a_ws[idx];
    float w  = outp[idx];
    hh = fmaf(av, hh, w);
    outp[idx] = hh;
    idx += D_;
  }
}

// ---------------------------------------------------------------------------
extern "C" void kernel_launch(void* const* d_in, const int* in_sizes, int n_in,
                              void* d_out, int out_size, void* d_ws, size_t ws_size,
                              hipStream_t stream) {
  const float* x       = (const float*)d_in[0];
  const float* ig_w    = (const float*)d_in[1];
  const float* ig_b    = (const float*)d_in[2];
  const float* ag_w    = (const float*)d_in[3];
  const float* ag_b    = (const float*)d_in[4];
  const float* a_param = (const float*)d_in[5];
  float* out = (float*)d_out;

  char* ws = (char*)d_ws;
  float*  a_ws = (float*)ws;                               // 128 MB: a[b][t][d]
  __bf16* wt   = (__bf16*)(ws + 134217728);                // 512 KB: bf16 W^T
  float*  P    = (float*)(ws + 134217728 + 524288);        // 256 KB
  float*  S    = (float*)(ws + 134217728 + 524288 + 262144);
  float*  C    = (float*)(ws + 134217728 + 524288 + 524288);

  rglru_prep_w<<<1024, 256, 0, stream>>>(ig_w, ag_w, wt);
  rglru_gates <<<4096, 256, 0, stream>>>(x, ig_b, ag_b, a_param, wt, a_ws, out);
  rglru_scan1 <<<256,  256, 0, stream>>>(a_ws, out, P, S);
  rglru_scan2 <<<32,   256, 0, stream>>>(P, S, C);
  rglru_scan3 <<<256,  256, 0, stream>>>(a_ws, C, out);
}
